// acLSTM_56178172231725
// MI455X (gfx1250) — matrix operand order repaired and checked by reading, not compiled, not run on hardware
//
#include <hip/hip_runtime.h>
#include <hip/hip_bf16.h>
#include <math.h>

#define IN_F   171
#define HID    1024
#define OUT_F  171
#define BATCH  128
#define SEQ_T  200
#define KX1    192   // layer-1 input K padded to multiple of 32
#define NDEC   176   // decoder N padded to multiple of 16

typedef __attribute__((ext_vector_type(16))) __bf16 bf16x16;
typedef __attribute__((ext_vector_type(8)))  float  f32x8;

union FragAB {
    bf16x16 v;
    uint4   q[2];
};

#if __has_builtin(__builtin_amdgcn_sched_barrier)
#define SCHED_FENCE() __builtin_amdgcn_sched_barrier(0)
#else
#define SCHED_FENCE() asm volatile("" ::: "memory")
#endif

__device__ __forceinline__ float sigmoidf_fast(float x) {
    return 1.0f / (1.0f + __expf(-x));
}

// Byte offset of a __shared__ object within the workgroup's LDS allocation.
__device__ __forceinline__ unsigned lds_offset_of(const void* p) {
    return (unsigned)(unsigned long long)(__attribute__((address_space(3))) const void*)p;
}

// CDNA5 async global->LDS copy (16B per lane), tracked by ASYNCcnt.
__device__ __forceinline__ void async_ld_b128(unsigned lds_byte_addr,
                                              unsigned long long gaddr) {
    asm volatile("global_load_async_to_lds_b128 %0, %1, off"
                 :: "v"(lds_byte_addr), "v"(gaddr)
                 : "memory");
}

__device__ __forceinline__ void wait_async0() {
    asm volatile("s_wait_asynccnt 0x0" ::: "memory");
}

// Async loads complete in order: ASYNCcnt <= 4 means the oldest chunk's four
// per-wave async instructions have landed while the next chunk stays in flight.
__device__ __forceinline__ void wait_async_le4() {
    asm volatile("s_wait_asynccnt 0x4" ::: "memory");
}

// ---------------------------------------------------------------------------
// fp32 -> bf16 weight conversion with zero padding (rows and cols)
// ---------------------------------------------------------------------------
__global__ void convert_pad_kernel(__bf16* __restrict__ dst, const float* __restrict__ src,
                                   int dstRows, int dstCols, int srcRows, int srcCols) {
    int total = dstRows * dstCols;
    for (int i = blockIdx.x * blockDim.x + threadIdx.x; i < total;
         i += gridDim.x * blockDim.x) {
        int r = i / dstCols;
        int c = i - r * dstCols;
        float v = (r < srcRows && c < srcCols) ? src[r * srcCols + c] : 0.0f;
        dst[i] = (__bf16)v;
    }
}

__global__ void bias_combine_kernel(float* __restrict__ dst, const float* __restrict__ a,
                                    const float* __restrict__ b, int n, int dstN) {
    int i = blockIdx.x * blockDim.x + threadIdx.x;
    if (i < dstN) {
        float v = 0.0f;
        if (i < n) {
            v = a[i];
            if (b) v += b[i];
        }
        dst[i] = v;
    }
}

__global__ void zero_kernel(float* __restrict__ p, int n) {
    for (int i = blockIdx.x * blockDim.x + threadIdx.x; i < n;
         i += gridDim.x * blockDim.x) {
        p[i] = 0.0f;
    }
}

// ---------------------------------------------------------------------------
// Build the layer-1 input frame in bf16: teacher-forced real_seq[:,t,:] or the
// model's own previous output d_out[:,t-1,:], zero-padded 171 -> 192.
// ---------------------------------------------------------------------------
__global__ void prep_x1_kernel(__bf16* __restrict__ x1,
                               const float* __restrict__ real_seq,
                               const float* __restrict__ prev_out,
                               const int* __restrict__ cnum,
                               const int* __restrict__ gnum, int t) {
    int i = blockIdx.x * blockDim.x + threadIdx.x;
    if (i >= BATCH * KX1) return;
    int m = i / KX1;
    int k = i - m * KX1;
    int cn = cnum[0], gn = gnum[0];
    int period = cn + gn;
    bool use_gt = (period <= 0) || ((t % period) < gn);
    float v = 0.0f;
    if (k < IN_F) {
        if (use_gt) {
            v = real_seq[(m * SEQ_T + t) * IN_F + k];
        } else if (t > 0) {
            v = prev_out[(m * SEQ_T + (t - 1)) * OUT_F + k];
        }
    }
    x1[i] = (__bf16)v;
}

// ---------------------------------------------------------------------------
// One LSTM layer, one timestep.
//   gates[128, 4H] = x @ Wx^T + h_prev @ Wh^T + bias; fused LSTM elementwise.
// grid.x = HID/32; block j owns hidden slice [32j, 32j+32) and computes all 4
// gate tiles for that slice. 256 threads = 8 waves; wave w owns batch rows
// [16w, 16w+16). A/B tiles triple-buffered in LDS via async global->LDS
// (ASYNCcnt), pipeline depth 2, one barrier per K-chunk. A sched fence keeps
// all 8 B fragments in distinct registers so the 8 WMMAs per chunk issue
// back-to-back after a single s_wait_dscnt (and no WMMA->VALU WAR nops).
// ---------------------------------------------------------------------------
__global__ void __launch_bounds__(256)
lstm_layer_kernel(const __bf16* __restrict__ x, int xK,
                  const __bf16* __restrict__ hprev,
                  const __bf16* __restrict__ Wx,
                  const __bf16* __restrict__ Wh,
                  const float* __restrict__ bias,
                  float* __restrict__ c,
                  __bf16* __restrict__ hnew) {
    __shared__ __align__(16) __bf16 Abuf[3][BATCH * 32];  // 3 x 8 KB
    __shared__ __align__(16) __bf16 Bbuf[3][128 * 32];    // 3 x 8 KB

    const int tid  = threadIdx.x;
    const int lane = tid & 31;
    const int wave = tid >> 5;
    const int js   = blockIdx.x * 32;   // hidden-slice base
    const int asel = lane >> 4;         // 0: lanes 0-15, 1: lanes 16-31

    f32x8 acc[8];
#pragma unroll
    for (int i = 0; i < 8; ++i)
#pragma unroll
        for (int e = 0; e < 8; ++e) acc[i][e] = 0.0f;

    const int nx = xK / 32;
    const int nchunks = nx + HID / 32;

    // per-thread staging coordinates (each thread moves 32B of A and 32B of B)
    const int sr = tid >> 1;       // 0..127
    const int sh = tid & 1;        // which 16B half-pair
    const int bgate = sr >> 5, bcol = sr & 31;

    const unsigned ldsA = lds_offset_of(&Abuf[0][0]);
    const unsigned ldsB = lds_offset_of(&Bbuf[0][0]);

    auto issue_chunk = [&](int idx, int buf) {
        const __bf16* A;
        const __bf16* W;
        int K, k0;
        if (idx < nx) { A = x;     W = Wx; K = xK;  k0 = idx * 32; }
        else          { A = hprev; W = Wh; K = HID; k0 = (idx - nx) * 32; }
        unsigned aL = ldsA + (unsigned)buf * (BATCH * 32 * 2) + sr * 64 + sh * 32;
        unsigned long long aG =
            (unsigned long long)(uintptr_t)(A + (size_t)sr * K + k0 + sh * 16);
        async_ld_b128(aL, aG);
        async_ld_b128(aL + 16, aG + 16);
        unsigned bL = ldsB + (unsigned)buf * (128 * 32 * 2) + sr * 64 + sh * 32;
        unsigned long long bG = (unsigned long long)(uintptr_t)(
            W + (size_t)(bgate * HID + js + bcol) * K + k0 + sh * 16);
        async_ld_b128(bL, bG);
        async_ld_b128(bL + 16, bG + 16);
    };

    issue_chunk(0, 0);
    issue_chunk(1, 1);

    int bufCur = 0;   // buffer consumed this iteration
    int bufIss = 2;   // buffer refilled this iteration (chunk i+2)

    for (int i = 0; i < nchunks; ++i) {
        if (i + 1 < nchunks) wait_async_le4();  // chunk i landed, i+1 in flight
        else                 wait_async0();     // last chunk: drain fully
        __syncthreads();                        // all waves' writes visible;
                                                // also fences re-use of bufIss
        if (i + 2 < nchunks) issue_chunk(i + 2, bufIss);

        // A fragment: 16x32 bf16 for this wave's 16 batch rows.
        // lanes 0-15 hold K 0-7 / 16-23, lanes 16-31 hold K 8-15 / 24-31.
        FragAB a;
        {
            int row = wave * 16 + (lane & 15);
            const __bf16* base = &Abuf[bufCur][row * 32];
            a.q[0] = *(const uint4*)(base + asel * 8);
            a.q[1] = *(const uint4*)(base + 16 + asel * 8);
        }
        // Batch all 8 B fragments into (distinct) registers...
        FragAB bfr[8];
#pragma unroll
        for (int g = 0; g < 4; ++g) {
#pragma unroll
            for (int nt = 0; nt < 2; ++nt) {
                int brow = g * 32 + nt * 16 + (lane & 15);
                const uint4* bp = (const uint4*)(&Bbuf[bufCur][brow * 32] + asel * 8);
                bfr[g * 2 + nt].q[0] = bp[0];
                bfr[g * 2 + nt].q[1] = bp[2];
            }
        }
        // ...then stream the WMMAs; the fence stops the scheduler from sinking
        // each load to just before its consumer (which would recycle one reg
        // set and serialize every WMMA behind a dscnt-0 wait).
        SCHED_FENCE();
#pragma unroll
        for (int j = 0; j < 8; ++j) {
            acc[j] = __builtin_amdgcn_wmma_f32_16x16x32_bf16(
                false, a.v, false, bfr[j].v, (short)0, acc[j], false, false);
        }

        bufCur = (bufCur == 2) ? 0 : bufCur + 1;
        bufIss = (bufIss == 2) ? 0 : bufIss + 1;
    }

    // ---- fused LSTM pointwise on accumulators ----
    // C/D layout: VGPR e, lanes 0-15 -> M = e, lanes 16-31 -> M = e+8; N = lane&15.
#pragma unroll
    for (int nt = 0; nt < 2; ++nt) {
        int jg = js + nt * 16 + (lane & 15);
        float bi = bias[jg];
        float bf = bias[HID + jg];
        float bg = bias[2 * HID + jg];
        float bo = bias[3 * HID + jg];
#pragma unroll
        for (int e = 0; e < 8; ++e) {
            int m = wave * 16 + e + asel * 8;
            float iv = sigmoidf_fast(acc[0 * 2 + nt][e] + bi);
            float fv = sigmoidf_fast(acc[1 * 2 + nt][e] + bf);
            float gv = tanhf(acc[2 * 2 + nt][e] + bg);
            float ov = sigmoidf_fast(acc[3 * 2 + nt][e] + bo);
            int idx = m * HID + jg;
            float cn = fv * c[idx] + iv * gv;
            c[idx] = cn;
            hnew[idx] = (__bf16)(ov * tanhf(cn));
        }
    }
}

// ---------------------------------------------------------------------------
// Decoder: out[:, t, :] = h2 @ Wdec^T + b_dec.  M=128, N=176(pad), K=1024.
// ---------------------------------------------------------------------------
__global__ void __launch_bounds__(256)
decoder_kernel(const __bf16* __restrict__ h2,
               const __bf16* __restrict__ Wdec,
               const float* __restrict__ bdec,
               float* __restrict__ out, int t) {
    const int tid  = threadIdx.x;
    const int lane = tid & 31;
    const int wave = tid >> 5;
    const int asel = lane >> 4;
    const int n0   = blockIdx.x * 16;

    f32x8 acc;
#pragma unroll
    for (int e = 0; e < 8; ++e) acc[e] = 0.0f;

    for (int k0 = 0; k0 < HID; k0 += 32) {
        FragAB a, b;
        int row = wave * 16 + (lane & 15);
        const __bf16* abase = h2 + row * HID + k0;
        a.q[0] = *(const uint4*)(abase + asel * 8);
        a.q[1] = *(const uint4*)(abase + 16 + asel * 8);

        int n = n0 + (lane & 15);
        const uint4* bp = (const uint4*)(Wdec + (size_t)n * HID + k0 + asel * 8);
        b.q[0] = bp[0];
        b.q[1] = bp[2];

        acc = __builtin_amdgcn_wmma_f32_16x16x32_bf16(
            false, a.v, false, b.v, (short)0, acc, false, false);
    }

    int n = n0 + (lane & 15);
    if (n < OUT_F) {
        float bv = bdec[n];
#pragma unroll
        for (int e = 0; e < 8; ++e) {
            int m = wave * 16 + e + asel * 8;
            out[(m * SEQ_T + t) * OUT_F + n] = acc[e] + bv;
        }
    }
}

// ---------------------------------------------------------------------------
extern "C" void kernel_launch(void* const* d_in, const int* in_sizes, int n_in,
                              void* d_out, int out_size, void* d_ws, size_t ws_size,
                              hipStream_t stream) {
    (void)in_sizes; (void)n_in; (void)out_size; (void)ws_size;

    const float* real_seq = (const float*)d_in[0];
    const int*   cnum     = (const int*)d_in[1];
    const int*   gnum     = (const int*)d_in[2];
    const float* W_ih1 = (const float*)d_in[3];
    const float* W_hh1 = (const float*)d_in[4];
    const float* b_ih1 = (const float*)d_in[5];
    const float* b_hh1 = (const float*)d_in[6];
    const float* W_ih2 = (const float*)d_in[7];
    const float* W_hh2 = (const float*)d_in[8];
    const float* b_ih2 = (const float*)d_in[9];
    const float* b_hh2 = (const float*)d_in[10];
    const float* W_ih3 = (const float*)d_in[11];
    const float* W_hh3 = (const float*)d_in[12];
    const float* b_ih3 = (const float*)d_in[13];
    const float* b_hh3 = (const float*)d_in[14];
    const float* W_dec = (const float*)d_in[15];
    const float* b_dec = (const float*)d_in[16];

    // ---- carve workspace (256B-aligned regions) ----
    uint8_t* ws = (uint8_t*)d_ws;
    size_t off = 0;
    auto alloc = [&](size_t bytes) -> uint8_t* {
        uint8_t* p = ws + off;
        off += (bytes + 255) & ~(size_t)255;
        return p;
    };
    __bf16* Wx1  = (__bf16*)alloc((size_t)4 * HID * KX1 * 2);
    __bf16* Wh1  = (__bf16*)alloc((size_t)4 * HID * HID * 2);
    __bf16* Wx2  = (__bf16*)alloc((size_t)4 * HID * HID * 2);
    __bf16* Wh2  = (__bf16*)alloc((size_t)4 * HID * HID * 2);
    __bf16* Wx3  = (__bf16*)alloc((size_t)4 * HID * HID * 2);
    __bf16* Wh3  = (__bf16*)alloc((size_t)4 * HID * HID * 2);
    __bf16* Wdec = (__bf16*)alloc((size_t)NDEC * HID * 2);
    float*  b1   = (float*)alloc((size_t)4 * HID * 4);
    float*  b2   = (float*)alloc((size_t)4 * HID * 4);
    float*  b3   = (float*)alloc((size_t)4 * HID * 4);
    float*  bdec = (float*)alloc((size_t)NDEC * 4);
    __bf16* hbuf[3][2];
    float*  cbuf[3];
    for (int l = 0; l < 3; ++l) {
        hbuf[l][0] = (__bf16*)alloc((size_t)BATCH * HID * 2);
        hbuf[l][1] = (__bf16*)alloc((size_t)BATCH * HID * 2);
        cbuf[l]    = (float*)alloc((size_t)BATCH * HID * 4);
    }
    __bf16* x1 = (__bf16*)alloc((size_t)BATCH * KX1 * 2);

    auto grid_for = [](int total) { return (total + 255) / 256; };

    // ---- one-time weight prep (re-done every call: deterministic) ----
    convert_pad_kernel<<<grid_for(4 * HID * KX1), 256, 0, stream>>>(
        Wx1, W_ih1, 4 * HID, KX1, 4 * HID, IN_F);
    convert_pad_kernel<<<grid_for(4 * HID * HID), 256, 0, stream>>>(
        Wh1, W_hh1, 4 * HID, HID, 4 * HID, HID);
    convert_pad_kernel<<<grid_for(4 * HID * HID), 256, 0, stream>>>(
        Wx2, W_ih2, 4 * HID, HID, 4 * HID, HID);
    convert_pad_kernel<<<grid_for(4 * HID * HID), 256, 0, stream>>>(
        Wh2, W_hh2, 4 * HID, HID, 4 * HID, HID);
    convert_pad_kernel<<<grid_for(4 * HID * HID), 256, 0, stream>>>(
        Wx3, W_ih3, 4 * HID, HID, 4 * HID, HID);
    convert_pad_kernel<<<grid_for(4 * HID * HID), 256, 0, stream>>>(
        Wh3, W_hh3, 4 * HID, HID, 4 * HID, HID);
    convert_pad_kernel<<<grid_for(NDEC * HID), 256, 0, stream>>>(
        Wdec, W_dec, NDEC, HID, OUT_F, HID);

    bias_combine_kernel<<<grid_for(4 * HID), 256, 0, stream>>>(b1, b_ih1, b_hh1, 4 * HID, 4 * HID);
    bias_combine_kernel<<<grid_for(4 * HID), 256, 0, stream>>>(b2, b_ih2, b_hh2, 4 * HID, 4 * HID);
    bias_combine_kernel<<<grid_for(4 * HID), 256, 0, stream>>>(b3, b_ih3, b_hh3, 4 * HID, 4 * HID);
    bias_combine_kernel<<<grid_for(NDEC), 256, 0, stream>>>(bdec, b_dec, nullptr, OUT_F, NDEC);

    // ---- zero recurrent state every call ----
    for (int l = 0; l < 3; ++l) {
        zero_kernel<<<grid_for(BATCH * HID / 2), 256, 0, stream>>>(
            (float*)hbuf[l][0], BATCH * HID / 2);
        zero_kernel<<<grid_for(BATCH * HID / 2), 256, 0, stream>>>(
            (float*)hbuf[l][1], BATCH * HID / 2);
        zero_kernel<<<grid_for(BATCH * HID), 256, 0, stream>>>(cbuf[l], BATCH * HID);
    }

    // ---- sequential time loop ----
    for (int t = 0; t < SEQ_T; ++t) {
        prep_x1_kernel<<<grid_for(BATCH * KX1), 256, 0, stream>>>(
            x1, real_seq, (const float*)d_out, cnum, gnum, t);
        int cur = t & 1, nxt = cur ^ 1;
        lstm_layer_kernel<<<HID / 32, 256, 0, stream>>>(
            x1, KX1, hbuf[0][cur], Wx1, Wh1, b1, cbuf[0], hbuf[0][nxt]);
        lstm_layer_kernel<<<HID / 32, 256, 0, stream>>>(
            hbuf[0][nxt], HID, hbuf[1][cur], Wx2, Wh2, b2, cbuf[1], hbuf[1][nxt]);
        lstm_layer_kernel<<<HID / 32, 256, 0, stream>>>(
            hbuf[1][nxt], HID, hbuf[2][cur], Wx3, Wh3, b3, cbuf[2], hbuf[2][nxt]);
        decoder_kernel<<<NDEC / 16, 256, 0, stream>>>(
            hbuf[2][nxt], Wdec, bdec, (float*)d_out, t);
    }
}
